// Transformer_Conv_7189775254147
// MI455X (gfx1250) — matrix operand order repaired and checked by reading, not compiled, not run on hardware
//
#include <hip/hip_runtime.h>
#include <hip/hip_bf16.h>

typedef __bf16 bf16;
typedef bf16  bf16x16 __attribute__((ext_vector_type(16)));
typedef bf16  bf16x8  __attribute__((ext_vector_type(8)));
typedef float f32x8   __attribute__((ext_vector_type(8)));
typedef int   i32x4   __attribute__((ext_vector_type(4)));

#define DIM   768
#define HEADS 12
#define DHEAD 64
#define MLP   3072
#define BATCH 8
#define SEQ   1024
#define ROWS  (BATCH * SEQ)   // 8192
#define QKVW  (3 * DIM)       // 2304

// ---------------------------------------------------------------------------
// CDNA5 async global->LDS path (guarded; falls back to b128 load + ds_store).
// Builtin signature (probe-confirmed): (int4 AS1*, int4 AS3*, imm off, imm cpol)
// ---------------------------------------------------------------------------
#if defined(__AMDGCN__) && __has_builtin(__builtin_amdgcn_global_load_async_to_lds_b128) && \
    __has_builtin(__builtin_amdgcn_s_wait_asynccnt)
#define USE_ASYNC_LDS 1
#define AS1 __attribute__((address_space(1)))
#define AS3 __attribute__((address_space(3)))
#else
#define USE_ASYNC_LDS 0
#endif

__device__ inline void copy16_g2lds(const bf16* g, bf16* l) {
#if USE_ASYNC_LDS
  __builtin_amdgcn_global_load_async_to_lds_b128((AS1 i32x4*)g, (AS3 i32x4*)l, 0, 0);
#else
  *(uint4*)l = *(const uint4*)g;
#endif
}
__device__ inline void async_wait() {
#if USE_ASYNC_LDS
  __builtin_amdgcn_s_wait_asynccnt(0);
#endif
}

// ---------------------------------------------------------------------------
// WMMA fragment helpers (layouts per CDNA5 ISA 7.12.2, wave32)
// A 16x32 bf16: lane m=lane&15, half=lane>>4; e<8 -> k=half*8+e, e>=8 -> 16+half*8+(e-8)
// B 32x16 bf16 stored [n][k] (B^T row-major): lane n=lane&15; e<8 -> k=half*8+e, e>=8 -> 16+half*8+(e-8)
// C/D 16x16 f32: lane n=lane&15; vgpr i -> row = i + (lane>>4)*8
// ---------------------------------------------------------------------------
__device__ inline bf16x16 frag_a(const bf16* p, int ld, int lane) {
  int m = lane & 15, h = lane >> 4;
  const bf16* q = p + m * ld + h * 8;
  bf16x16 a;
#pragma unroll
  for (int e = 0; e < 8; ++e) { a[e] = q[e]; a[e + 8] = q[16 + e]; }
  return a;
}
__device__ inline bf16x16 frag_b(const bf16* p, int ld, int lane) {
  int n = lane & 15, h = lane >> 4;
  const bf16* q = p + n * ld + h * 8;
  bf16x16 b;
#pragma unroll
  for (int e = 0; e < 8; ++e) { b[e] = q[e]; b[e + 8] = q[16 + e]; }
  return b;
}
__device__ inline f32x8 wmma_bf16(bf16x16 a, bf16x16 b, f32x8 c) {
  return __builtin_amdgcn_wmma_f32_16x16x32_bf16(false, a, false, b,
                                                 (short)0, c, false, false);
}

// ---------------------------------------------------------------------------
// LayerNorm (f32 in, OutT out): one block per row, 256 threads x 3 cols
// ---------------------------------------------------------------------------
template <typename OutT>
__global__ __launch_bounds__(256) void ln_kernel(const float* __restrict__ x,
                                                 const float* __restrict__ g,
                                                 const float* __restrict__ b,
                                                 OutT* __restrict__ out) {
  int row = blockIdx.x, tid = threadIdx.x;
  const float* xr = x + (size_t)row * DIM;
  float v0 = xr[tid], v1 = xr[tid + 256], v2 = xr[tid + 512];
  float s = v0 + v1 + v2;
#pragma unroll
  for (int m = 1; m < 32; m <<= 1) s += __shfl_xor(s, m, 32);
  __shared__ float red[8];
  int lane = tid & 31, wv = tid >> 5;
  if (lane == 0) red[wv] = s;
  __syncthreads();
  float tot = 0.f;
#pragma unroll
  for (int i = 0; i < 8; ++i) tot += red[i];
  float mu = tot * (1.0f / DIM);
  float d0 = v0 - mu, d1 = v1 - mu, d2 = v2 - mu;
  float sq = d0 * d0 + d1 * d1 + d2 * d2;
#pragma unroll
  for (int m = 1; m < 32; m <<= 1) sq += __shfl_xor(sq, m, 32);
  __syncthreads();
  if (lane == 0) red[wv] = sq;
  __syncthreads();
  float tsq = 0.f;
#pragma unroll
  for (int i = 0; i < 8; ++i) tsq += red[i];
  float rstd = rsqrtf(tsq * (1.0f / DIM) + 1e-5f);
  OutT* orow = out + (size_t)row * DIM;
  orow[tid]       = (OutT)(d0 * rstd * g[tid]       + b[tid]);
  orow[tid + 256] = (OutT)(d1 * rstd * g[tid + 256] + b[tid + 256]);
  orow[tid + 512] = (OutT)(d2 * rstd * g[tid + 512] + b[tid + 512]);
}

// ---------------------------------------------------------------------------
// Weight transpose+convert: Wt[n][k] = (bf16)(W[k][n] * (n<scale_cols ? 1/8 : 1))
// (scale_cols=768 folds the attention 1/sqrt(d) into the Q projection)
// ---------------------------------------------------------------------------
__global__ __launch_bounds__(256) void transpose_bf16_kernel(const float* __restrict__ W,
                                                             bf16* __restrict__ Wt,
                                                             int K, int Nn,
                                                             int scale_cols) {
  __shared__ float tile[32][33];
  int n0 = blockIdx.x * 32, k0 = blockIdx.y * 32;
  int tx = threadIdx.x & 31, ty = threadIdx.x >> 5;
#pragma unroll
  for (int i = ty; i < 32; i += 8)
    tile[i][tx] = W[(size_t)(k0 + i) * Nn + n0 + tx];
  __syncthreads();
#pragma unroll
  for (int i = ty; i < 32; i += 8) {
    float sc = (n0 + i < scale_cols) ? 0.125f : 1.0f;
    Wt[(size_t)(n0 + i) * K + k0 + tx] = (bf16)(tile[tx][i] * sc);
  }
}

// ---------------------------------------------------------------------------
// bf16-WMMA GEMM, software-pipelined (double-buffered LDS, async copies for
// tile k+1 overlap WMMAs of tile k). 128x256 block tile, 8 waves (2M x 4N),
// wave 64x64 = 4x4 WMMA.
// MODE 0: C(bf16) = A@Wt^T ; MODE 1: C(f32) = ..+bias ; MODE 2: C(f32) += ..+bias
// ---------------------------------------------------------------------------
#define GLDA 40  // 32 + 8 pad: row stride 80B, conflict-free 16B frag reads

__device__ inline void gemm_stage(const bf16* __restrict__ A,
                                  const bf16* __restrict__ Wt,
                                  int m0, int n0, int kt, int K, int tid,
                                  bf16 (&As)[128][GLDA], bf16 (&Bs)[256][GLDA]) {
#pragma unroll
  for (int it = 0; it < 2; ++it) {
    int lin = (tid + it * 256) * 8;
    int r = lin >> 5, c = lin & 31;
    const bf16* src = A + (size_t)(m0 + r) * K + kt + c;
    copy16_g2lds(src, &As[r][c]);
    if (kt + 32 < K) __builtin_prefetch(src + 32, 0, 1);
  }
#pragma unroll
  for (int it = 0; it < 4; ++it) {
    int lin = (tid + it * 256) * 8;
    int n = lin >> 5, c = lin & 31;
    const bf16* src = Wt + (size_t)(n0 + n) * K + kt + c;
    copy16_g2lds(src, &Bs[n][c]);
    if (kt + 32 < K) __builtin_prefetch(src + 32, 0, 1);
  }
}

template <int MODE>
__global__ __launch_bounds__(256) void gemm_kernel(const bf16* __restrict__ A,
                                                   const bf16* __restrict__ Wt,
                                                   const float* __restrict__ bias,
                                                   void* __restrict__ Cv,
                                                   int Nn, int K) {
  __shared__ bf16 As[2][128][GLDA];
  __shared__ bf16 Bs[2][256][GLDA];
  int tid = threadIdx.x, lane = tid & 31, wv = tid >> 5;
  int wm = wv & 1, wn = wv >> 1;
  int m0 = blockIdx.y * 128, n0 = blockIdx.x * 256;

  f32x8 z = {0.f, 0.f, 0.f, 0.f, 0.f, 0.f, 0.f, 0.f};
  f32x8 acc[4][4];
#pragma unroll
  for (int mi = 0; mi < 4; ++mi)
#pragma unroll
    for (int ni = 0; ni < 4; ++ni) acc[mi][ni] = z;

  // prologue: stage tile 0 into buffer 0
  gemm_stage(A, Wt, m0, n0, 0, K, tid, As[0], Bs[0]);

  for (int kt = 0; kt < K; kt += 32) {
    int cur = (kt >> 5) & 1;
    async_wait();      // my tile-kt copies have landed
    __syncthreads();   // everyone's landed; prev-iter reads of other buf done
    if (kt + 32 < K)   // overlap next tile's DMA with this tile's WMMAs
      gemm_stage(A, Wt, m0, n0, kt + 32, K, tid, As[cur ^ 1], Bs[cur ^ 1]);

    bf16x16 af[4], bf_[4];
#pragma unroll
    for (int mi = 0; mi < 4; ++mi)
      af[mi] = frag_a(&As[cur][wm * 64 + mi * 16][0], GLDA, lane);
#pragma unroll
    for (int ni = 0; ni < 4; ++ni)
      bf_[ni] = frag_b(&Bs[cur][wn * 64 + ni * 16][0], GLDA, lane);
#pragma unroll
    for (int mi = 0; mi < 4; ++mi)
#pragma unroll
      for (int ni = 0; ni < 4; ++ni)
        acc[mi][ni] = wmma_bf16(af[mi], bf_[ni], acc[mi][ni]);
  }

  int nl = lane & 15, hf = lane >> 4;
  if (MODE == 0) {
    bf16* C = (bf16*)Cv;
#pragma unroll
    for (int mi = 0; mi < 4; ++mi)
#pragma unroll
      for (int ni = 0; ni < 4; ++ni) {
        int col = n0 + wn * 64 + ni * 16 + nl;
#pragma unroll
        for (int i = 0; i < 8; ++i) {
          int row = m0 + wm * 64 + mi * 16 + hf * 8 + i;
          C[(size_t)row * Nn + col] = (bf16)acc[mi][ni][i];
        }
      }
  } else {
    float* C = (float*)Cv;
#pragma unroll
    for (int mi = 0; mi < 4; ++mi)
#pragma unroll
      for (int ni = 0; ni < 4; ++ni) {
        int col = n0 + wn * 64 + ni * 16 + nl;
        float bv = bias[col];
#pragma unroll
        for (int i = 0; i < 8; ++i) {
          int row = m0 + wm * 64 + mi * 16 + hf * 8 + i;
          size_t idx = (size_t)row * Nn + col;
          float v = acc[mi][ni][i] + bv;
          if (MODE == 2) v += C[idx];
          C[idx] = v;
        }
      }
  }
}

// ---------------------------------------------------------------------------
// Flash attention on bf16 QKV (Q pre-scaled by 1/sqrt(d) via wqkvT).
// One (b,h,qblock) per workgroup, Q block = 128 rows (8 waves x 16),
// KV blocks of 64, online softmax in registers.
// ---------------------------------------------------------------------------
__global__ __launch_bounds__(256) void attn_kernel(const bf16* __restrict__ qkv,
                                                   bf16* __restrict__ out) {
  __shared__ bf16 Qs[128][72];
  __shared__ bf16 Ks[64][72];
  __shared__ bf16 Vt[64][72];   // [d][key]
  __shared__ bf16 Ps[128][72];

  int tid = threadIdx.x, lane = tid & 31, wv = tid >> 5;
  int qb = blockIdx.x * 128;
  int h = blockIdx.y, b = blockIdx.z;

#pragma unroll
  for (int it = 0; it < 4; ++it) {
    int lin = (tid + it * 256) * 8;
    int r = lin >> 6, d = lin & 63;
    copy16_g2lds(qkv + (size_t)(b * SEQ + qb + r) * QKVW + h * 64 + d, &Qs[r][d]);
  }

  int wrow = wv * 16;
  int nl = lane & 15, hf = lane >> 4;
  f32x8 z = {0.f, 0.f, 0.f, 0.f, 0.f, 0.f, 0.f, 0.f};
  f32x8 oacc[4];
#pragma unroll
  for (int t = 0; t < 4; ++t) oacc[t] = z;
  float mprev[8], lsum[8];
#pragma unroll
  for (int i = 0; i < 8; ++i) { mprev[i] = -1e30f; lsum[i] = 0.f; }

  for (int kb = 0; kb < SEQ; kb += 64) {
    __syncthreads();
#pragma unroll
    for (int it = 0; it < 2; ++it) {
      int lin = (tid + it * 256) * 8;
      int r = lin >> 6, d = lin & 63;
      const bf16* base = qkv + (size_t)(b * SEQ + kb + r) * QKVW + h * 64 + d;
      copy16_g2lds(base + DIM, &Ks[r][d]);          // K natural [key][d]
      bf16x8 vv = *(const bf16x8*)(base + 2 * DIM); // V transposed scatter
#pragma unroll
      for (int j = 0; j < 8; ++j) Vt[d + j][r] = vv[j];
    }
    async_wait();
    __syncthreads();

    // S = Q.K^T (16 q-rows x 64 keys per wave)
    f32x8 sacc[4];
#pragma unroll
    for (int ni = 0; ni < 4; ++ni) sacc[ni] = z;
#pragma unroll
    for (int ks = 0; ks < 2; ++ks) {
      bf16x16 aq = frag_a(&Qs[wrow][ks * 32], 72, lane);
#pragma unroll
      for (int ni = 0; ni < 4; ++ni) {
        bf16x16 bk = frag_b(&Ks[ni * 16][ks * 32], 72, lane);
        sacc[ni] = wmma_bf16(aq, bk, sacc[ni]);
      }
    }

    // online softmax (group-of-16 xor reductions give per-row stats)
    float pm[8], rs[8];
#pragma unroll
    for (int i = 0; i < 8; ++i)
      pm[i] = fmaxf(fmaxf(sacc[0][i], sacc[1][i]), fmaxf(sacc[2][i], sacc[3][i]));
#pragma unroll
    for (int m = 1; m < 16; m <<= 1)
#pragma unroll
      for (int i = 0; i < 8; ++i) pm[i] = fmaxf(pm[i], __shfl_xor(pm[i], m, 32));
#pragma unroll
    for (int i = 0; i < 8; ++i) {
      float nm = fmaxf(mprev[i], pm[i]);
      float alpha = __expf(mprev[i] - nm);
      mprev[i] = nm;
      float r = 0.f;
#pragma unroll
      for (int ni = 0; ni < 4; ++ni) {
        float p = __expf(sacc[ni][i] - nm);
        sacc[ni][i] = p;
        r += p;
      }
      rs[i] = r;
      lsum[i] *= alpha;
#pragma unroll
      for (int t = 0; t < 4; ++t) oacc[t][i] *= alpha;
    }
#pragma unroll
    for (int m = 1; m < 16; m <<= 1)
#pragma unroll
      for (int i = 0; i < 8; ++i) rs[i] += __shfl_xor(rs[i], m, 32);
#pragma unroll
    for (int i = 0; i < 8; ++i) lsum[i] += rs[i];

    // P (C-layout regs) -> LDS A-layout, own rows only
#pragma unroll
    for (int ni = 0; ni < 4; ++ni)
#pragma unroll
      for (int i = 0; i < 8; ++i)
        Ps[wrow + hf * 8 + i][ni * 16 + nl] = (bf16)sacc[ni][i];

    // O += P.V
#pragma unroll
    for (int kk = 0; kk < 2; ++kk) {
      bf16x16 ap = frag_a(&Ps[wrow][kk * 32], 72, lane);
#pragma unroll
      for (int t = 0; t < 4; ++t) {
        bf16x16 bv = frag_b(&Vt[t * 16][kk * 32], 72, lane);
        oacc[t] = wmma_bf16(ap, bv, oacc[t]);
      }
    }
  }

  float invl[8];
#pragma unroll
  for (int i = 0; i < 8; ++i) invl[i] = 1.0f / lsum[i];
#pragma unroll
  for (int t = 0; t < 4; ++t)
#pragma unroll
    for (int i = 0; i < 8; ++i) {
      int row = qb + wrow + hf * 8 + i;
      out[(size_t)(b * SEQ + row) * DIM + h * 64 + t * 16 + nl] =
          (bf16)(oacc[t][i] * invl[i]);
    }
}

// ---------------------------------------------------------------------------
// 3x3 depthwise conv (32x32 spatial, zero pad) + bias + exact GELU, f32->bf16
// ---------------------------------------------------------------------------
__global__ __launch_bounds__(256) void dwconv_gelu_kernel(const float* __restrict__ hin,
                                                          const float* __restrict__ kern,
                                                          const float* __restrict__ kb,
                                                          bf16* __restrict__ hout) {
  int c = blockIdx.x * 256 + threadIdx.x;
  int n = blockIdx.y, b = blockIdx.z;
  int y = n >> 5, x = n & 31;
  const float* kc = kern + (size_t)c * 9;
  float s = 0.f;
#pragma unroll
  for (int dy = 0; dy < 3; ++dy) {
    int yy = y + dy - 1;
    if (yy < 0 || yy > 31) continue;
#pragma unroll
    for (int dx = 0; dx < 3; ++dx) {
      int xx = x + dx - 1;
      if (xx < 0 || xx > 31) continue;
      s += hin[(size_t)(b * SEQ + (yy << 5) + xx) * MLP + c] * kc[dy * 3 + dx];
    }
  }
  s += kb[c];
  float gv = 0.5f * s * (1.0f + erff(s * 0.70710678118f));
  hout[(size_t)(b * SEQ + n) * MLP + c] = (bf16)gv;
}

// ---------------------------------------------------------------------------
extern "C" void kernel_launch(void* const* d_in, const int* in_sizes, int n_in,
                              void* d_out, int out_size, void* d_ws, size_t ws_size,
                              hipStream_t stream) {
  (void)in_sizes; (void)n_in; (void)out_size; (void)ws_size;
  const float* x_in = (const float*)d_in[0];
  const float* ln1g = (const float*)d_in[1];
  const float* ln1b = (const float*)d_in[2];
  const float* wqkv = (const float*)d_in[3];
  const float* wout = (const float*)d_in[4];
  const float* bout = (const float*)d_in[5];
  const float* ln2g = (const float*)d_in[6];
  const float* ln2b = (const float*)d_in[7];
  const float* w1   = (const float*)d_in[8];
  const float* b1   = (const float*)d_in[9];
  const float* dwk  = (const float*)d_in[10];
  const float* dwb  = (const float*)d_in[11];
  const float* w2   = (const float*)d_in[12];
  const float* b2   = (const float*)d_in[13];
  const float* lnfg = (const float*)d_in[14];
  const float* lnfb = (const float*)d_in[15];

  char* p = (char*)d_ws;
  float* xbuf = (float*)p; p += (size_t)ROWS * DIM * 4;
  bf16* hln   = (bf16*)p;  p += (size_t)ROWS * DIM * 2;   // LN out / attn out
  bf16* qkv   = (bf16*)p;  p += (size_t)ROWS * QKVW * 2;
  float* h1   = (float*)p; p += (size_t)ROWS * MLP * 4;
  bf16* h2    = (bf16*)p;  p += (size_t)ROWS * MLP * 2;
  bf16* wqkvT = (bf16*)p;  p += (size_t)4 * QKVW * DIM * 2;
  bf16* woutT = (bf16*)p;  p += (size_t)4 * DIM * DIM * 2;
  bf16* w1T   = (bf16*)p;  p += (size_t)4 * MLP * DIM * 2;
  bf16* w2T   = (bf16*)p;  p += (size_t)4 * DIM * MLP * 2;

  (void)hipMemcpyAsync(xbuf, x_in, (size_t)ROWS * DIM * sizeof(float),
                       hipMemcpyDeviceToDevice, stream);

  // one-time weight transpose+convert to bf16 Wt[N][K];
  // Q columns of wqkv scaled by 1/sqrt(DHEAD)
  for (int L = 0; L < 4; ++L) {
    transpose_bf16_kernel<<<dim3(QKVW / 32, DIM / 32), 256, 0, stream>>>(
        wqkv + (size_t)L * DIM * QKVW, wqkvT + (size_t)L * QKVW * DIM, DIM, QKVW, DIM);
    transpose_bf16_kernel<<<dim3(DIM / 32, DIM / 32), 256, 0, stream>>>(
        wout + (size_t)L * DIM * DIM, woutT + (size_t)L * DIM * DIM, DIM, DIM, 0);
    transpose_bf16_kernel<<<dim3(MLP / 32, DIM / 32), 256, 0, stream>>>(
        w1 + (size_t)L * DIM * MLP, w1T + (size_t)L * MLP * DIM, DIM, MLP, 0);
    transpose_bf16_kernel<<<dim3(DIM / 32, MLP / 32), 256, 0, stream>>>(
        w2 + (size_t)L * MLP * DIM, w2T + (size_t)L * DIM * MLP, MLP, DIM, 0);
  }

  for (int L = 0; L < 4; ++L) {
    // --- attention block ---
    ln_kernel<bf16><<<ROWS, 256, 0, stream>>>(xbuf, ln1g + L * DIM, ln1b + L * DIM, hln);
    gemm_kernel<0><<<dim3(QKVW / 256, ROWS / 128), 256, 0, stream>>>(
        hln, wqkvT + (size_t)L * QKVW * DIM, nullptr, qkv, QKVW, DIM);
    attn_kernel<<<dim3(SEQ / 128, HEADS, BATCH), 256, 0, stream>>>(qkv, hln);
    gemm_kernel<2><<<dim3(DIM / 256, ROWS / 128), 256, 0, stream>>>(
        hln, woutT + (size_t)L * DIM * DIM, bout + L * DIM, xbuf, DIM, DIM);
    // --- conv-MLP block ---
    ln_kernel<bf16><<<ROWS, 256, 0, stream>>>(xbuf, ln2g + L * DIM, ln2b + L * DIM, hln);
    gemm_kernel<1><<<dim3(MLP / 256, ROWS / 128), 256, 0, stream>>>(
        hln, w1T + (size_t)L * MLP * DIM, b1 + (size_t)L * MLP, h1, MLP, DIM);
    dwconv_gelu_kernel<<<dim3(MLP / 256, SEQ, BATCH), 256, 0, stream>>>(
        h1, dwk + (size_t)L * MLP * 9, dwb + (size_t)L * MLP, h2);
    gemm_kernel<2><<<dim3(DIM / 256, ROWS / 128), 256, 0, stream>>>(
        h2, w2T + (size_t)L * DIM * MLP, b2 + L * DIM, xbuf, DIM, MLP);
  }
  ln_kernel<float><<<ROWS, 256, 0, stream>>>(xbuf, lnfg, lnfb, (float*)d_out);
}
